// Seq2seq_attn_63213328663200
// MI455X (gfx1250) — matrix operand order repaired and checked by reading, not compiled, not run on hardware
//
#include <hip/hip_runtime.h>
#include <math.h>

// ---------------- types ----------------
typedef __attribute__((ext_vector_type(16))) __bf16 v16bf;
typedef __attribute__((ext_vector_type(8)))  float  v8f;

__device__ __forceinline__ unsigned short f2bf(float f) {
    unsigned int u = __float_as_uint(f);
    u += 0x7FFFu + ((u >> 16) & 1u);           // round-to-nearest-even
    return (unsigned short)(u >> 16);
}

// ---------------- weight convert + transpose: W[N,K] f32 -> Wt[K,N] bf16 -----
__global__ void conv_tr_bf16(const float* __restrict__ W,
                             unsigned short* __restrict__ Wt,
                             int N, int K) {
    long long idx = (long long)blockIdx.x * blockDim.x + threadIdx.x;
    if (idx >= (long long)N * K) return;
    int n = (int)(idx / K);
    int k = (int)(idx % K);
    Wt[(size_t)k * N + n] = f2bf(W[idx]);
}

// ---------------- embeddings ----------------
__global__ void embed_enc(const int* __restrict__ x, const float* __restrict__ emb,
                          unsigned short* __restrict__ out) {
    int idx = blockIdx.x * 256 + threadIdx.x;      // B*S*H = 2,097,152
    int bs = idx >> 10, h = idx & 1023;
    int tok = x[bs];
    out[idx] = f2bf(emb[(size_t)tok * 1024 + h]);
}

__global__ void embed_tok(const int* __restrict__ tok, const float* __restrict__ emb,
                          unsigned short* __restrict__ out) {
    int idx = blockIdx.x * 256 + threadIdx.x;      // B*H = 32768
    int b = idx >> 10, h = idx & 1023;
    out[idx] = f2bf(emb[(size_t)tok[b] * 1024 + h]);
}

// ---------------- fills / copies ----------------
__global__ void zero_f32(float* p, int n)          { int i = blockIdx.x*256+threadIdx.x; if (i < n) p[i] = 0.f; }
__global__ void zero_bf16(unsigned short* p, int n){ int i = blockIdx.x*256+threadIdx.x; if (i < n) p[i] = 0;  }
__global__ void zero_i32(int* p, int n)            { int i = blockIdx.x*256+threadIdx.x; if (i < n) p[i] = 0;  }
__global__ void copy_f32(const float* __restrict__ s, float* __restrict__ d, int n) {
    int i = blockIdx.x*256+threadIdx.x; if (i < n) d[i] = s[i];
}

// ---------------- WMMA GEMM: C[M,N] = A[M,K](bf16) @ Bt[K,N](bf16) + bias ----
// Block: 256 threads = 8 waves; block tile 32(M) x 256(N); wave tile 32x32
// (2x2 wmma, register reuse of A and B fragments).  B panel [32K x 256N] is
// double-buffered through LDS with global_load_async_to_lds_b128 (ASYNCcnt).
// grid = (N/256, M/32). K multiple of 32. A row-major lda; Bt row-major ldb=N.
__global__ void __launch_bounds__(256)
gemm_bf16_wmma(const unsigned short* __restrict__ A, int lda,
               const unsigned short* __restrict__ Bt, int ldb,
               const float* __restrict__ bias,
               float* __restrict__ C, long long ldc, int K) {
    // 264-halves row pitch (528 B = 132 dwords) to break LDS bank conflicts
    __shared__ unsigned short smem[2][32][264];

    const int wv   = threadIdx.x >> 5;       // 0..7
    const int lane = threadIdx.x & 31;
    const int m0   = blockIdx.y * 32;
    const int nblk = blockIdx.x * 256;
    const int nw   = nblk + wv * 32;         // this wave's 32 N columns

    // A-matrix 16x32 bf16 layout (ISA 7.12.2): lane = m + 16*khalf,
    // halves 0..7 -> K = off+0..7, halves 8..15 -> K = off+16..23, off=khalf*8
    const int mrow  = lane & 15;
    const int khalf = lane >> 4;
    const unsigned short* arow0 = A + (size_t)(m0 + mrow) * lda + khalf * 8;
    const unsigned short* arow1 = arow0 + (size_t)16 * lda;

    // async panel mapping: 256 threads cover 32(K) x 256(N) bf16 = 16 KB,
    // 4 x b128 per thread -> 4 async instructions per wave per panel
    const int pr = threadIdx.x >> 3;         // k-row 0..31
    const int pc = threadIdx.x & 7;          // 8 threads per row

    v8f acc00 = {}, acc01 = {}, acc10 = {}, acc11 = {};

    // prologue: issue panel 0 into buffer 0
    {
        const unsigned short* g = Bt + (size_t)pr * ldb + nblk + pc * 32;
#pragma unroll
        for (int j = 0; j < 4; ++j) {
            unsigned laddr = (unsigned)(size_t)&smem[0][pr][pc * 32 + j * 8];
            unsigned long long gaddr = (unsigned long long)(size_t)(g + j * 8);
            asm volatile("global_load_async_to_lds_b128 %0, %1, off"
                         :: "v"(laddr), "v"(gaddr) : "memory");
        }
    }

    for (int k0 = 0; k0 < K; k0 += 32) {
        const int cur = (k0 >> 5) & 1;
        if (k0 + 32 < K) {
            // issue next panel into the other buffer, then wait for current
            const unsigned short* g = Bt + (size_t)(k0 + 32 + pr) * ldb + nblk + pc * 32;
#pragma unroll
            for (int j = 0; j < 4; ++j) {
                unsigned laddr = (unsigned)(size_t)&smem[cur ^ 1][pr][pc * 32 + j * 8];
                unsigned long long gaddr = (unsigned long long)(size_t)(g + j * 8);
                asm volatile("global_load_async_to_lds_b128 %0, %1, off"
                             :: "v"(laddr), "v"(gaddr) : "memory");
            }
            asm volatile("s_wait_asynccnt 0x4" ::: "memory");
        } else {
            asm volatile("s_wait_asynccnt 0x0" ::: "memory");
        }
        __syncthreads();

        union U { uint4 u[2]; v16bf v; } a0, a1, b0, b1;
        a0.u[0] = *(const uint4*)(arow0 + k0);
        a0.u[1] = *(const uint4*)(arow0 + k0 + 16);
        a1.u[0] = *(const uint4*)(arow1 + k0);
        a1.u[1] = *(const uint4*)(arow1 + k0 + 16);
        // B-matrix 32x16: lane holds column N=lane%16, K in the same order as A
        const unsigned short* br = &smem[cur][0][(wv << 5) + mrow];
#pragma unroll
        for (int s = 0; s < 16; ++s) {
            const int kk = (s < 8) ? (khalf * 8 + s) : (16 + khalf * 8 + (s - 8));
            ((unsigned short*)&b0)[s] = br[kk * 264];
            ((unsigned short*)&b1)[s] = br[kk * 264 + 16];
        }

        acc00 = __builtin_amdgcn_wmma_f32_16x16x32_bf16(false, a0.v, false, b0.v, (short)0, acc00, false, false);
        acc01 = __builtin_amdgcn_wmma_f32_16x16x32_bf16(false, a0.v, false, b1.v, (short)0, acc01, false, false);
        acc10 = __builtin_amdgcn_wmma_f32_16x16x32_bf16(false, a1.v, false, b0.v, (short)0, acc10, false, false);
        acc11 = __builtin_amdgcn_wmma_f32_16x16x32_bf16(false, a1.v, false, b1.v, (short)0, acc11, false, false);
        __syncthreads();
    }

    const int   ncol0 = nw + (lane & 15);
    const int   mb    = m0 + ((lane >> 4) << 3);   // +8 for lanes 16..31
    const float bb0   = bias ? bias[ncol0]      : 0.f;
    const float bb1   = bias ? bias[ncol0 + 16] : 0.f;
    const size_t ldcs = (size_t)ldc;
#pragma unroll
    for (int r = 0; r < 8; ++r) {
        C[(size_t)(mb + r)      * ldcs + ncol0]      = acc00[r] + bb0;
        C[(size_t)(mb + r)      * ldcs + ncol0 + 16] = acc01[r] + bb1;
        C[(size_t)(mb + 16 + r) * ldcs + ncol0]      = acc10[r] + bb0;
        C[(size_t)(mb + 16 + r) * ldcs + ncol0 + 16] = acc11[r] + bb1;
    }
}

// ---------------- GRU gate fusion ----------------
// gi row index = b*rows_per_b + row_off (pitch 3072); gh row = b.
__global__ void gru_gate(const float* __restrict__ gi, const float* __restrict__ gh,
                         int rows_per_b, int row_off,
                         float* __restrict__ h, unsigned short* __restrict__ hbf,
                         float* __restrict__ seq_f32, unsigned short* __restrict__ seq_bf16) {
    int idx = blockIdx.x * 256 + threadIdx.x;      // B*H = 32768
    int b = idx >> 10, hh = idx & 1023;
    size_t grow = ((size_t)b * rows_per_b + row_off) * 3072;
    size_t hrow = (size_t)b * 3072;
    float ir = gi[grow + hh], iz = gi[grow + 1024 + hh], inn = gi[grow + 2048 + hh];
    float hr = gh[hrow + hh], hz = gh[hrow + 1024 + hh], hn = gh[hrow + 2048 + hh];
    float r = 1.f / (1.f + __expf(-(ir + hr)));
    float z = 1.f / (1.f + __expf(-(iz + hz)));
    float n = tanhf(inn + r * hn);
    float hold = h[idx];
    float hp = (1.f - z) * n + z * hold;
    h[idx]   = hp;
    hbf[idx] = f2bf(hp);
    size_t sidx = ((size_t)b * rows_per_b + row_off) * 1024 + hh;
    if (seq_f32)  seq_f32[sidx]  = hp;
    if (seq_bf16) seq_bf16[sidx] = f2bf(hp);
}

// ---------------- attention scores: scores[b,s] = q[b,:] . enc[b,s,:] --------
__global__ void attn_scores(const float* __restrict__ q, const float* __restrict__ enc,
                            float* __restrict__ scores) {
    int g = blockIdx.x * 8 + (threadIdx.x >> 5);   // 2048 waves total
    int lane = threadIdx.x & 31;
    int b = g >> 6, s = g & 63;
    const float* qp = q + (size_t)b * 1024;
    const float* ep = enc + ((size_t)b * 64 + s) * 1024;
    float sum = 0.f;
    for (int h = lane; h < 1024; h += 32) sum += qp[h] * ep[h];
    for (int off = 16; off; off >>= 1) sum += __shfl_xor(sum, off, 32);
    if (lane == 0) scores[b * 64 + s] = sum;
}

// ---------------- softmax over S=64 + context ----------------
__global__ void softmax_ctx(const float* __restrict__ scores, const float* __restrict__ enc,
                            float* __restrict__ attn_out,
                            float* __restrict__ ctx, unsigned short* __restrict__ ctxbf) {
    __shared__ float sa[64];
    __shared__ float sm, ss;
    int b = blockIdx.x, tid = threadIdx.x;
    if (tid < 64) sa[tid] = scores[b * 64 + tid];
    __syncthreads();
    if (tid == 0) { float m = sa[0]; for (int i = 1; i < 64; ++i) m = fmaxf(m, sa[i]); sm = m; }
    __syncthreads();
    if (tid < 64) sa[tid] = __expf(sa[tid] - sm);
    __syncthreads();
    if (tid == 0) { float t = 0.f; for (int i = 0; i < 64; ++i) t += sa[i]; ss = t; }
    __syncthreads();
    if (tid < 64) { sa[tid] /= ss; attn_out[b * 64 + tid] = sa[tid]; }
    __syncthreads();
    const float* ep = enc + (size_t)b * 64 * 1024;
    for (int h = tid; h < 1024; h += 256) {
        float a = 0.f;
        for (int s = 0; s < 64; ++s) a += sa[s] * ep[(size_t)s * 1024 + h];
        ctx[b * 1024 + h]   = a;
        ctxbf[b * 1024 + h] = f2bf(a);
    }
}

// ---------------- argmax over V=32000 (first max wins) ----------------
__global__ void argmax_v(const float* __restrict__ logits, long long row_stride,
                         int* __restrict__ tok) {
    __shared__ float bv[256];
    __shared__ int   bi[256];
    int b = blockIdx.x, tid = threadIdx.x;
    const float* lp = logits + (size_t)b * row_stride;
    float best = -3.4e38f; int besti = 0x7fffffff;
    for (int v = tid; v < 32000; v += 256) {
        float xv = lp[v];
        if (xv > best) { best = xv; besti = v; }
    }
    bv[tid] = best; bi[tid] = besti;
    __syncthreads();
    for (int st = 128; st; st >>= 1) {
        if (tid < st) {
            if (bv[tid + st] > bv[tid] ||
                (bv[tid + st] == bv[tid] && bi[tid + st] < bi[tid])) {
                bv[tid] = bv[tid + st]; bi[tid] = bi[tid + st];
            }
        }
        __syncthreads();
    }
    if (tid == 0) tok[b] = bi[0];
}

// =====================================================================
extern "C" void kernel_launch(void* const* d_in, const int* in_sizes, int n_in,
                              void* d_out, int out_size, void* d_ws, size_t ws_size,
                              hipStream_t stream) {
    (void)in_sizes; (void)n_in; (void)out_size; (void)ws_size;
    const int*   x       = (const int*)  d_in[0];
    const float* fr_emb  = (const float*)d_in[1];
    const float* en_emb  = (const float*)d_in[2];
    const float* enc_Wih = (const float*)d_in[3];
    const float* enc_Whh = (const float*)d_in[4];
    const float* enc_bih = (const float*)d_in[5];
    const float* enc_bhh = (const float*)d_in[6];
    const float* dec_Wih = (const float*)d_in[7];
    const float* dec_Whh = (const float*)d_in[8];
    const float* dec_bih = (const float*)d_in[9];
    const float* dec_bhh = (const float*)d_in[10];
    const float* Wq      = (const float*)d_in[11];
    const float* bq      = (const float*)d_in[12];
    const float* Wfc     = (const float*)d_in[13];
    const float* bfc     = (const float*)d_in[14];
    float* out = (float*)d_out;

    const int H = 1024, S = 64, B = 32, T = 64, V = 32000;
    const size_t LW = (size_t)H * 3 * H;            // transposed per-layer weight elems

    // ---- workspace carve-out ----
    char* ws = (char*)d_ws;
    size_t cur = 0;
    auto alloc = [&](size_t bytes) -> char* {
        char* p = ws + cur;
        cur = (cur + bytes + 255) & ~(size_t)255;
        return p;
    };
    unsigned short* encWihT = (unsigned short*)alloc(2 * LW * 2);
    unsigned short* encWhhT = (unsigned short*)alloc(2 * LW * 2);
    unsigned short* decWihT = (unsigned short*)alloc(2 * LW * 2);
    unsigned short* decWhhT = (unsigned short*)alloc(2 * LW * 2);
    unsigned short* WqT     = (unsigned short*)alloc((size_t)H * H * 2);
    unsigned short* WfcT    = (unsigned short*)alloc((size_t)H * V * 2);
    unsigned short* embX    = (unsigned short*)alloc((size_t)B * S * H * 2);
    unsigned short* ys0     = (unsigned short*)alloc((size_t)B * S * H * 2);
    float*          enc_out = (float*)        alloc((size_t)B * S * H * 4);
    float*          gi_big  = (float*)        alloc((size_t)B * S * 3 * H * 4);
    float*          gh      = (float*)        alloc((size_t)B * 3 * H * 4);
    float*          gi_dec  = (float*)        alloc((size_t)B * 3 * H * 4);
    float*          h_dec   = (float*)        alloc((size_t)2 * B * H * 4);
    unsigned short* h_decbf = (unsigned short*)alloc((size_t)2 * B * H * 2);
    unsigned short* x_dec   = (unsigned short*)alloc((size_t)B * H * 2);
    float*          qbuf    = (float*)        alloc((size_t)B * H * 4);
    float*          scoresb = (float*)        alloc((size_t)B * S * 4);
    float*          attnws  = (float*)        alloc((size_t)B * S * 4);
    float*          ctx     = (float*)        alloc((size_t)B * H * 4);
    unsigned short* ctxbf   = (unsigned short*)alloc((size_t)B * H * 2);
    int*            tok     = (int*)          alloc(B * 4);

    auto cdiv = [](long long a, long long b) { return (unsigned)((a + b - 1) / b); };
    dim3 blk(256);

    // ---- weight conversion: f32 [N,K] -> bf16 [K,N] (whole set fits in L2) --
    const unsigned NB3H = cdiv((long long)3 * H * H, 256);
    for (int l = 0; l < 2; ++l) {
        conv_tr_bf16<<<NB3H, blk, 0, stream>>>(enc_Wih + (size_t)l * 3 * H * H, encWihT + (size_t)l * LW, 3 * H, H);
        conv_tr_bf16<<<NB3H, blk, 0, stream>>>(enc_Whh + (size_t)l * 3 * H * H, encWhhT + (size_t)l * LW, 3 * H, H);
        conv_tr_bf16<<<NB3H, blk, 0, stream>>>(dec_Wih + (size_t)l * 3 * H * H, decWihT + (size_t)l * LW, 3 * H, H);
        conv_tr_bf16<<<NB3H, blk, 0, stream>>>(dec_Whh + (size_t)l * 3 * H * H, decWhhT + (size_t)l * LW, 3 * H, H);
    }
    conv_tr_bf16<<<cdiv((long long)H * H, 256), blk, 0, stream>>>(Wq, WqT, H, H);
    conv_tr_bf16<<<cdiv((long long)V * H, 256), blk, 0, stream>>>(Wfc, WfcT, V, H);

    // ---- encoder ----
    embed_enc<<<cdiv((long long)B * S * H, 256), blk, 0, stream>>>(x, fr_emb, embX);
    zero_f32 <<<cdiv(2 * B * H, 256), blk, 0, stream>>>(h_dec, 2 * B * H);
    zero_bf16<<<cdiv(2 * B * H, 256), blk, 0, stream>>>(h_decbf, 2 * B * H);

    const unsigned short* layer_in = embX;
    for (int l = 0; l < 2; ++l) {
        // all-timestep input-side GEMM: [B*S, H] @ [H, 3H]
        gemm_bf16_wmma<<<dim3(3 * H / 256, B * S / 32), blk, 0, stream>>>(
            layer_in, H, encWihT + (size_t)l * LW, 3 * H,
            enc_bih + (size_t)l * 3 * H, gi_big, 3 * H, H);
        for (int t = 0; t < S; ++t) {
            gemm_bf16_wmma<<<dim3(3 * H / 256, 1), blk, 0, stream>>>(
                h_decbf + (size_t)l * B * H, H, encWhhT + (size_t)l * LW, 3 * H,
                enc_bhh + (size_t)l * 3 * H, gh, 3 * H, H);
            gru_gate<<<cdiv(B * H, 256), blk, 0, stream>>>(
                gi_big, gh, S, t,
                h_dec + (size_t)l * B * H, h_decbf + (size_t)l * B * H,
                (l == 1) ? enc_out : nullptr, (l == 0) ? ys0 : nullptr);
        }
        layer_in = ys0;
    }

    // ---- decoder (greedy, with attention) ----
    zero_i32<<<1, blk, 0, stream>>>(tok, B);
    for (int t = 0; t < T; ++t) {
        embed_tok<<<cdiv(B * H, 256), blk, 0, stream>>>(tok, en_emb, x_dec);
        // layer 0
        gemm_bf16_wmma<<<dim3(3 * H / 256, 1), blk, 0, stream>>>(
            x_dec, H, decWihT, 3 * H, dec_bih, gi_dec, 3 * H, H);
        gemm_bf16_wmma<<<dim3(3 * H / 256, 1), blk, 0, stream>>>(
            h_decbf, H, decWhhT, 3 * H, dec_bhh, gh, 3 * H, H);
        gru_gate<<<cdiv(B * H, 256), blk, 0, stream>>>(
            gi_dec, gh, 1, 0, h_dec, h_decbf, nullptr, nullptr);
        // layer 1 (input = layer-0 hidden just updated)
        gemm_bf16_wmma<<<dim3(3 * H / 256, 1), blk, 0, stream>>>(
            h_decbf, H, decWihT + LW, 3 * H, dec_bih + 3 * H, gi_dec, 3 * H, H);
        gemm_bf16_wmma<<<dim3(3 * H / 256, 1), blk, 0, stream>>>(
            h_decbf + (size_t)B * H, H, decWhhT + LW, 3 * H, dec_bhh + 3 * H, gh, 3 * H, H);
        gru_gate<<<cdiv(B * H, 256), blk, 0, stream>>>(
            gi_dec, gh, 1, 0, h_dec + (size_t)B * H, h_decbf + (size_t)B * H, nullptr, nullptr);
        // attention
        gemm_bf16_wmma<<<dim3(H / 256, 1), blk, 0, stream>>>(
            h_decbf + (size_t)B * H, H, WqT, H, bq, qbuf, H, H);
        attn_scores<<<B * S / 8, blk, 0, stream>>>(qbuf, enc_out, scoresb);
        float* attn_ptr = (t == T - 1)
                        ? out + (size_t)B * T * V + (size_t)2 * B * H   // attn_last slice
                        : attnws;
        softmax_ctx<<<B, blk, 0, stream>>>(scoresb, enc_out, attn_ptr, ctx, ctxbf);
        // logits -> d_out[b, t, :] ; row stride across b is T*V
        gemm_bf16_wmma<<<dim3(V / 256, 1), blk, 0, stream>>>(
            ctxbf, H, WfcT, V, bfc, out + (size_t)t * V, (long long)T * V, H);
        argmax_v<<<B, blk, 0, stream>>>(out + (size_t)t * V, (long long)T * V, tok);
    }

    // ---- final hidden hT -> d_out ----
    copy_f32<<<cdiv(2 * B * H, 256), blk, 0, stream>>>(h_dec, out + (size_t)B * T * V, 2 * B * H);
}
